// Encoder_86388972192062
// MI455X (gfx1250) — matrix operand order repaired and checked by reading, not compiled, not run on hardware
//
#include <hip/hip_runtime.h>
#include <math.h>

typedef __bf16 bf16;
typedef __attribute__((ext_vector_type(16))) __bf16 v16bf;
typedef __attribute__((ext_vector_type(8)))  __bf16 v8bf;
typedef __attribute__((ext_vector_type(8)))  float  v8f;

#define NODES 50000
#define EDGES 600000
#define DIM   128
#define EDIM  16
#define MT    80     // M rows per GEMM block: 50000 = 625 * 80 (exact)

// ---------------------------------------------------------------- utilities

__global__ void f32_to_bf16_k(const float* __restrict__ in, bf16* __restrict__ out, int n) {
    int i = blockIdx.x * blockDim.x + threadIdx.x;
    if (i < n) out[i] = (bf16)in[i];
}

// leaky_relu(0.01) then convert to bf16 (feeds layer-2 GEMMs)
__global__ void leaky_to_bf16_k(const float* __restrict__ in, bf16* __restrict__ out, int n) {
    int i = blockIdx.x * blockDim.x + threadIdx.x;
    if (i < n) {
        float v = in[i];
        out[i] = (bf16)(v > 0.f ? v : 0.01f * v);
    }
}

// W [128(in) x 128(out)] f32 -> Wt [128(out) x 128(in)] bf16 (transposed for
// contiguous B-fragment reads from LDS)
__global__ void convT_w_k(const float* __restrict__ W, bf16* __restrict__ Wt) {
    int i = blockIdx.x * blockDim.x + threadIdx.x;   // i = n*128 + k
    int n = i >> 7, k = i & 127;
    if (i < DIM * DIM) Wt[i] = (bf16)W[k * DIM + n];
}

__global__ void init_softmax_k(float* __restrict__ m, float* __restrict__ den, int n) {
    int i = blockIdx.x * blockDim.x + threadIdx.x;
    if (i < n) { m[i] = -__builtin_huge_valf(); den[i] = 0.f; }
}

// ---------------------------------------------------------------- WMMA GEMM
// out[N,128] = Xb[N,128](bf16) @ W(128,128) + bias ; Wt is W transposed bf16.
// Block = 256 threads = 8 waves; block owns an 80-row M tile; wave w owns the
// 16-col N tile n0=16w and 5 M sub-tiles -> 20 x v_wmma_f32_16x16x32_bf16,
// with each B fragment reused across 5 accumulators.
__global__ __launch_bounds__(256)
void gemm_wmma_k(const bf16* __restrict__ X, const bf16* __restrict__ Wt,
                 const float* __restrict__ bias, float* __restrict__ out) {
    __shared__ bf16 ldsX[MT * DIM];     // 20 KB
    __shared__ bf16 ldsW[DIM * DIM];    // 32 KB (transposed weights)

    const int    tid  = threadIdx.x;
    const size_t row0 = (size_t)blockIdx.x * MT;

    // cooperative loads: X tile = 80 contiguous rows (20480 B), Wt = 32768 B
    const uint4* xsrc = (const uint4*)(X + row0 * DIM);
#pragma unroll
    for (int i = 0; i < 5; ++i)
        ((uint4*)ldsX)[tid + 256 * i] = xsrc[tid + 256 * i];
    const uint4* wsrc = (const uint4*)Wt;
#pragma unroll
    for (int i = 0; i < 8; ++i)
        ((uint4*)ldsW)[tid + 256 * i] = wsrc[tid + 256 * i];
    __syncthreads();

    const int lane = tid & 31;
    const int wave = tid >> 5;
    const int half = lane >> 4;     // 0: lanes 0-15, 1: lanes 16-31
    const int lidx = lane & 15;
    const int n0   = wave * 16;

    const bf16* wrow = ldsW + (n0 + lidx) * DIM;       // B: N = lidx (Wt rows)

    v8f acc[5] = {};
#pragma unroll
    for (int kk = 0; kk < 4; ++kk) {
        // B 32x16 bf16: lane half h holds K = kk*32 + 16h + [0..15]
        v16bf b = *(const v16bf*)(wrow + kk * 32 + half * 16);
        const int kb = kk * 32 + half * 16;
#pragma unroll
        for (int s2 = 0; s2 < 5; ++s2) {
            // A 16x32 bf16: lane half h holds K = kk*32 + 16h + [0..15]
            const bf16* xrow = ldsX + (s2 * 16 + lidx) * DIM;
            v8bf alo = *(const v8bf*)(xrow + kb);
            v8bf ahi = *(const v8bf*)(xrow + kb + 8);
            v16bf a;
#pragma unroll
            for (int i = 0; i < 8; ++i) { a[i] = alo[i]; a[i + 8] = ahi[i]; }
            acc[s2] = __builtin_amdgcn_wmma_f32_16x16x32_bf16(
                          false, a, false, b, (short)0, acc[s2], false, false);
        }
    }

    const int   n  = n0 + lidx;
    const float bn = bias[n];
#pragma unroll
    for (int s2 = 0; s2 < 5; ++s2) {
#pragma unroll
        for (int r = 0; r < 8; ++r) {
            const size_t mrow = row0 + s2 * 16 + r + half * 8; // D: VGPR r -> M=r+8h
            out[mrow * DIM + n] = acc[s2][r] + bn;
        }
    }
}

// ---------------------------------------------------------------- edge passes
// Edge embedding e[col] = be[col] + sum_j ea[e,j]*We[j,col] is recomputed on
// the fly (64-256 FMA/edge) instead of materializing a 307 MB [E,128] buffer.

__device__ __forceinline__ void atomic_max_f32(float* addr, float v) {
    if (v >= 0.f) atomicMax((int*)addr, __float_as_int(v));
    else          atomicMin((unsigned int*)addr, (unsigned int)__float_as_int(v));
}

// Pass 1: score[e,h] = <q[dst,h,:], k[src,h,:]+e[h,:]>/sqrt(C); atomic max -> m
template <int H>
__global__ void edge_score_k(const int* __restrict__ src, const int* __restrict__ dst,
                             const float* __restrict__ ea,
                             const float* __restrict__ We, const float* __restrict__ be,
                             const float* __restrict__ Q, const float* __restrict__ K,
                             float* __restrict__ score, float* __restrict__ mbuf) {
    constexpr int C = DIM / H;
    long t = (long)blockIdx.x * blockDim.x + threadIdx.x;
    if (t >= (long)EDGES * H) return;
    int e = (int)(t / H);
    int h = (int)(t - (long)e * H);
    int s = src[e], d = dst[e];

    const float* qr = Q + (size_t)d * DIM + h * C;
    const float* kr = K + (size_t)s * DIM + h * C;
    __builtin_prefetch(qr, 0, 3);            // global_prefetch_b8: start gather
    __builtin_prefetch(kr, 0, 3);            // before the 16x16 FMA block

    float ec[EDIM];
    const float4* er4 = (const float4*)(ea + (size_t)e * EDIM);
#pragma unroll
    for (int j = 0; j < EDIM / 4; ++j) {
        float4 v = er4[j];
        ec[4 * j + 0] = v.x; ec[4 * j + 1] = v.y;
        ec[4 * j + 2] = v.z; ec[4 * j + 3] = v.w;
    }

    float acc = 0.f;
    for (int c = 0; c < C; ++c) {
        const int col = h * C + c;
        float ev = be[col];
#pragma unroll
        for (int j = 0; j < EDIM; ++j) ev += ec[j] * We[j * DIM + col];
        acc += qr[c] * (kr[c] + ev);
    }
    const float sc = acc * (1.0f / sqrtf((float)C));
    score[t] = sc;
    atomic_max_f32(mbuf + (size_t)d * H + h, sc);
}

// Pass 2: ex = exp(score - m[dst]) ; atomic add -> den
template <int H>
__global__ void edge_expden_k(const int* __restrict__ dst,
                              float* __restrict__ score,
                              const float* __restrict__ mbuf,
                              float* __restrict__ den) {
    long t = (long)blockIdx.x * blockDim.x + threadIdx.x;
    if (t >= (long)EDGES * H) return;
    int e = (int)(t / H);
    int h = (int)(t - (long)e * H);
    int d = dst[e];
    float ex = expf(score[t] - mbuf[(size_t)d * H + h]);
    score[t] = ex;                                   // overwrite with exp value
    atomicAdd(den + (size_t)d * H + h, ex);
}

// Pass 3: out[dst] += (v[src]+e) * ex/(den+eps). 1 wave / edge, 4 cols / lane.
template <int H>
__global__ __launch_bounds__(256)
void edge_message_k(const int* __restrict__ src, const int* __restrict__ dst,
                    const float* __restrict__ ea,
                    const float* __restrict__ We, const float* __restrict__ be,
                    const float* __restrict__ V,
                    const float* __restrict__ exbuf, const float* __restrict__ den,
                    float* __restrict__ out) {
    constexpr int C = DIM / H;
    const int e = blockIdx.x * 8 + (threadIdx.x >> 5);
    if (e >= EDGES) return;
    const int lane = threadIdx.x & 31;
    const int s = src[e], d = dst[e];

    const float* vrow = V + (size_t)s * DIM + lane * 4;   // 4 contiguous cols
    __builtin_prefetch(vrow, 0, 3);

    float ec[EDIM];
    const float4* er4 = (const float4*)(ea + (size_t)e * EDIM);
#pragma unroll
    for (int j = 0; j < EDIM / 4; ++j) {
        float4 v = er4[j];
        ec[4 * j + 0] = v.x; ec[4 * j + 1] = v.y;
        ec[4 * j + 2] = v.z; ec[4 * j + 3] = v.w;
    }

    const float4 v4 = *(const float4*)vrow;
    const float  vv[4] = {v4.x, v4.y, v4.z, v4.w};

#pragma unroll
    for (int u = 0; u < 4; ++u) {
        const int col = lane * 4 + u;
        const int h   = col / C;
        float ev = be[col];
#pragma unroll
        for (int j = 0; j < EDIM; ++j) ev += ec[j] * We[j * DIM + col];
        const float alpha = exbuf[(size_t)e * H + h] /
                            (den[(size_t)d * H + h] + 1e-16f);
        const float msg = (vv[u] + ev) * alpha;
        atomicAdd(out + (size_t)d * DIM + col, msg);
    }
}

// ---------------------------------------------------------------- launcher

extern "C" void kernel_launch(void* const* d_in, const int* in_sizes, int n_in,
                              void* d_out, int out_size, void* d_ws, size_t ws_size,
                              hipStream_t stream) {
    (void)in_sizes; (void)n_in; (void)out_size; (void)ws_size;
    const float* x   = (const float*)d_in[0];
    const int*   ei  = (const int*)  d_in[1];
    const float* ea  = (const float*)d_in[2];
    const float *Wq1 = (const float*)d_in[3],  *bq1 = (const float*)d_in[4];
    const float *Wk1 = (const float*)d_in[5],  *bk1 = (const float*)d_in[6];
    const float *Wv1 = (const float*)d_in[7],  *bv1 = (const float*)d_in[8];
    const float *We1 = (const float*)d_in[9],  *be1 = (const float*)d_in[10];
    const float *Ws1 = (const float*)d_in[11], *bs1 = (const float*)d_in[12];
    const float *Wq2 = (const float*)d_in[13], *bq2 = (const float*)d_in[14];
    const float *Wk2 = (const float*)d_in[15], *bk2 = (const float*)d_in[16];
    const float *Wv2 = (const float*)d_in[17], *bv2 = (const float*)d_in[18];
    const float *We2 = (const float*)d_in[19], *be2 = (const float*)d_in[20];
    const float *Ws2 = (const float*)d_in[21], *bs2 = (const float*)d_in[22];

    const int* src = ei;
    const int* dst = ei + EDGES;

    char*  base = (char*)d_ws;
    size_t off  = 0;
    auto take = [&](size_t bytes) -> char* {
        char* p = base + off;
        off += (bytes + 255) & ~(size_t)255;
        return p;
    };
    bf16*  xb = (bf16*) take((size_t)NODES * DIM * sizeof(bf16));
    float* qb = (float*)take((size_t)NODES * DIM * sizeof(float));
    float* kb = (float*)take((size_t)NODES * DIM * sizeof(float));
    float* vb = (float*)take((size_t)NODES * DIM * sizeof(float));
    float* h1 = (float*)take((size_t)NODES * DIM * sizeof(float));
    float* sc = (float*)take((size_t)EDGES * 8 * sizeof(float));
    float* mb = (float*)take((size_t)NODES * 8 * sizeof(float));
    float* db = (float*)take((size_t)NODES * 8 * sizeof(float));
    bf16*  wtq = (bf16*)take(DIM * DIM * sizeof(bf16));
    bf16*  wtk = (bf16*)take(DIM * DIM * sizeof(bf16));
    bf16*  wtv = (bf16*)take(DIM * DIM * sizeof(bf16));
    bf16*  wts = (bf16*)take(DIM * DIM * sizeof(bf16));
    float* out = (float*)d_out;

    const int ND   = NODES * DIM;                 // 6.4M
    const int gNE  = (ND + 255) / 256;
    const int gW   = (DIM * DIM + 255) / 256;     // 64
    const int gM   = NODES / MT;                  // 625 (exact)
    const long EH8 = (long)EDGES * 8;
    const int gE8  = (int)((EH8 + 255) / 256);
    const int gE1  = (EDGES + 255) / 256;
    const int gMsg = (EDGES + 7) / 8;

    // ---------------- layer 1 (H=8, C=16, concat) ----------------
    f32_to_bf16_k<<<gNE, 256, 0, stream>>>(x, xb, ND);
    convT_w_k<<<gW, 256, 0, stream>>>(Wq1, wtq);
    convT_w_k<<<gW, 256, 0, stream>>>(Wk1, wtk);
    convT_w_k<<<gW, 256, 0, stream>>>(Wv1, wtv);
    convT_w_k<<<gW, 256, 0, stream>>>(Ws1, wts);
    gemm_wmma_k<<<gM, 256, 0, stream>>>(xb, wtq, bq1, qb);
    gemm_wmma_k<<<gM, 256, 0, stream>>>(xb, wtk, bk1, kb);
    gemm_wmma_k<<<gM, 256, 0, stream>>>(xb, wtv, bv1, vb);
    gemm_wmma_k<<<gM, 256, 0, stream>>>(xb, wts, bs1, h1);  // skip preloads out
    init_softmax_k<<<(NODES * 8 + 255) / 256, 256, 0, stream>>>(mb, db, NODES * 8);
    edge_score_k<8><<<gE8, 256, 0, stream>>>(src, dst, ea, We1, be1, qb, kb, sc, mb);
    edge_expden_k<8><<<gE8, 256, 0, stream>>>(dst, sc, mb, db);
    edge_message_k<8><<<gMsg, 256, 0, stream>>>(src, dst, ea, We1, be1, vb, sc, db, h1);
    leaky_to_bf16_k<<<gNE, 256, 0, stream>>>(h1, xb, ND);

    // ---------------- layer 2 (H=1, C=128, mean==identity) ----------------
    convT_w_k<<<gW, 256, 0, stream>>>(Wq2, wtq);
    convT_w_k<<<gW, 256, 0, stream>>>(Wk2, wtk);
    convT_w_k<<<gW, 256, 0, stream>>>(Wv2, wtv);
    convT_w_k<<<gW, 256, 0, stream>>>(Ws2, wts);
    gemm_wmma_k<<<gM, 256, 0, stream>>>(xb, wtq, bq2, qb);
    gemm_wmma_k<<<gM, 256, 0, stream>>>(xb, wtk, bk2, kb);
    gemm_wmma_k<<<gM, 256, 0, stream>>>(xb, wtv, bv2, vb);
    gemm_wmma_k<<<gM, 256, 0, stream>>>(xb, wts, bs2, out); // skip preloads d_out
    init_softmax_k<<<(NODES + 255) / 256, 256, 0, stream>>>(mb, db, NODES);
    edge_score_k<1><<<gE1, 256, 0, stream>>>(src, dst, ea, We2, be2, qb, kb, sc, mb);
    edge_expden_k<1><<<gE1, 256, 0, stream>>>(dst, sc, mb, db);
    edge_message_k<1><<<gMsg, 256, 0, stream>>>(src, dst, ea, We2, be2, vb, sc, db, out);
}
